// AttentionLayer_515396076122
// MI455X (gfx1250) — matrix operand order repaired and checked by reading, not compiled, not run on hardware
//
#include <hip/hip_runtime.h>
#include <hip/hip_bf16.h>

// Flash-attention (Q=K=V=X), B=8, S=2048, H=512, fp32 I/O, bf16 WMMA compute.
// QT=64 queries per workgroup (4x less KV re-read traffic than QT=16),
// KT=32-key KV steps staged in LDS. 8 wave32 per WG; 150KB LDS -> 2 WG/WGP.

typedef __attribute__((ext_vector_type(16))) __bf16 v16bf;
typedef __attribute__((ext_vector_type(8)))  float  v8f;

#define B_  8
#define S_  2048
#define H_  512
#define QT  64        // queries per workgroup
#define KT  32        // keys per KV step

// Padded strides (element units) to spread LDS banks, keeping 16B alignment.
#define PQ  520       // Qbf row stride (u16) : 1040B = 260 dw == 4 mod 64
#define PK  520       // Kbf row stride (u16)
#define PV  40        // Vt  row stride (u16) :   80B =  20 dw (gcd 4 -> spread)
#define PS  36        // Sc  row stride (f32) :  144B =  36 dw (gcd 4 -> spread)

// LDS layout (bytes), all offsets 16B aligned
#define OFF_Q 0
#define OFF_K (OFF_Q + QT * PQ * 2)      // 66560
#define OFF_V (OFF_K + KT * PK * 2)      // 99840
#define OFF_S (OFF_V + H_ * PV * 2)      // 140800
#define SMEM_BYTES (OFF_S + QT * PS * 4) // 150016  (2 WGs fit in 320KB WGP LDS)

__device__ __forceinline__ unsigned short f2bf(float f) {
  unsigned int u = __float_as_uint(f);
  unsigned int r = u + 0x7FFFu + ((u >> 16) & 1u);   // round-to-nearest-even
  return (unsigned short)(r >> 16);
}

union AFrag {
  v16bf v;
  unsigned short h[16];
  uint4 q[2];
};

union U16x8 {
  uint4 q;
  unsigned short e[8];
};

extern "C" __global__ __launch_bounds__(256)
void fattn_bf16_wmma(const float* __restrict__ X, float* __restrict__ Out) {
  extern __shared__ char smem[];
  unsigned short* Qbf = (unsigned short*)(smem + OFF_Q); // [64][PQ]  bf16
  unsigned short* Kbf = (unsigned short*)(smem + OFF_K); // [32][PK]  bf16
  unsigned short* Vt  = (unsigned short*)(smem + OFF_V); // [512][PV] bf16 (transposed)
  float*          Sc  = (float*)(smem + OFF_S);          // [64][PS]  f32 scores

  const int tid  = threadIdx.x;
  const int lane = tid & 31;
  const int wv   = tid >> 5;                 // wave 0..7 -> h slice [64*wv, 64*wv+64)
  const int b    = blockIdx.x >> 5;          // / (S/QT = 32)
  const int q0   = (blockIdx.x & 31) * QT;
  const float* Xb = X + (size_t)b * S_ * H_;

  // ---- Stage Q tile: 64x512 f32 -> bf16 in LDS (coalesced, contiguous stores) ----
  for (int i = tid; i < QT * H_ / 4; i += 256) {
    int row = i >> 7;                        // 128 float4 per row
    int h   = (i & 127) << 2;
    float4 f = *(const float4*)(Xb + (size_t)(q0 + row) * H_ + h);
    unsigned int lo = (unsigned int)f2bf(f.x) | ((unsigned int)f2bf(f.y) << 16);
    unsigned int hi = (unsigned int)f2bf(f.z) | ((unsigned int)f2bf(f.w) << 16);
    *(uint2*)&Qbf[row * PQ + h] = make_uint2(lo, hi);
  }

  const v8f zero8 = {0.f,0.f,0.f,0.f,0.f,0.f,0.f,0.f};
  v8f acc[16];                                // [qh][t]: 64q x 64h f32 accumulator
  #pragma unroll
  for (int t = 0; t < 16; ++t) acc[t] = zero8;
  float m_run[4], l_run[4];                   // per query-half stats (row = lane&15)
  #pragma unroll
  for (int qh = 0; qh < 4; ++qh) { m_run[qh] = -__builtin_inff(); l_run[qh] = 0.f; }

  const int row = lane & 15;

  for (int kt = 0; kt < S_ / KT; ++kt) {
    const int k0 = kt * KT;
    __syncthreads();                          // previous-step LDS consumers done

    // ---- A1: global -> Kbf (row-major bf16); coalesced loads, contiguous stores ----
    for (int i = tid; i < KT * H_ / 4; i += 256) {
      int key = i >> 7;
      int h   = (i & 127) << 2;
      float4 f = *(const float4*)(Xb + (size_t)(k0 + key) * H_ + h);
      unsigned int lo = (unsigned int)f2bf(f.x) | ((unsigned int)f2bf(f.y) << 16);
      unsigned int hi = (unsigned int)f2bf(f.z) | ((unsigned int)f2bf(f.w) << 16);
      *(uint2*)&Kbf[key * PK + h] = make_uint2(lo, hi);
    }
    // L2 prefetch hint for next KV tile (global_prefetch_b8): 32 rows x 2KB
    if (kt + 1 < S_ / KT)
      __builtin_prefetch(Xb + (size_t)(k0 + KT + (tid & 31)) * H_ + ((tid >> 5) << 6), 0, 1);
    __syncthreads();

    // ---- A2: LDS transpose Kbf -> Vt; lanes walk keys => conflict-free stores ----
    for (int i = tid; i < KT * H_ / 8; i += 256) {
      int key = i & 31;
      int h   = (i >> 5) << 3;
      U16x8 d;
      d.q = *(const uint4*)&Kbf[key * PK + h];
      #pragma unroll
      for (int j = 0; j < 8; ++j)
        Vt[(h + j) * PV + key] = d.e[j];
    }
    __syncthreads();

    // ---- Scores: wave wv computes one 16x16 tile of the 64x32 score block ----
    {
      const int qi   = wv >> 1;                     // query 16-row tile 0..3
      const int ks   = wv & 1;                      // key 16-col slice 0..1
      const int rowA = (qi << 4) + row;
      const int kbA  = (lane >> 4) << 3;            // A frag K-split: 0 / 8
      const int keyB = (ks << 4) + (lane & 15);     // B frag column (key)
      const int kbB  = (lane >> 4) << 3;            // B frag K-split: 0 / 8
      v8f s = zero8;
      #pragma unroll
      for (int c = 0; c < H_ / 32; ++c) {
        const int h0 = c << 5;
        AFrag a, bb;
        a.q[0]  = *(const uint4*)&Qbf[rowA * PQ + h0 + kbA];
        a.q[1]  = *(const uint4*)&Qbf[rowA * PQ + h0 + kbA + 16];
        bb.q[0] = *(const uint4*)&Kbf[keyB * PK + h0 + kbB];
        bb.q[1] = *(const uint4*)&Kbf[keyB * PK + h0 + kbB + 16];
        s = __builtin_amdgcn_wmma_f32_16x16x32_bf16(false, a.v, false, bb.v,
                                                    (short)0, s, false, false);
      }
      const int hb = (lane >> 4) << 3;              // C layout: row = g + hb
      const int n  = lane & 15;
      #pragma unroll
      for (int g = 0; g < 8; ++g)
        Sc[((qi << 4) + g + hb) * PS + (ks << 4) + n] = s[g];
    }
    __syncthreads();

    // ---- Fused online-softmax + P@V, one query-half (16 rows) at a time.
    //      Lane owns row = lane&15 of each half; lane^16 covers the other
    //      K-split of the same row -> full-row stats via shfl_xor. ----
    {
      const int kbA = (lane >> 4) << 3;
      const int nc  = lane & 15;
      const int kst = (lane >> 4) << 3;
      const int hb  = (lane >> 4) << 3;
      #pragma unroll
      for (int qh = 0; qh < 4; ++qh) {
        const int base = ((qh << 4) + row) * PS + kbA;
        float4 s0 = *(const float4*)&Sc[base];          // keys kbA+0..3
        float4 s1 = *(const float4*)&Sc[base + 4];      // keys kbA+4..7
        float4 s2 = *(const float4*)&Sc[base + 16];     // keys kbA+16..19
        float4 s3 = *(const float4*)&Sc[base + 20];     // keys kbA+20..23
        float hmax = fmaxf(fmaxf(fmaxf(s0.x, s0.y), fmaxf(s0.z, s0.w)),
                           fmaxf(fmaxf(s1.x, s1.y), fmaxf(s1.z, s1.w)));
        hmax = fmaxf(hmax, fmaxf(fmaxf(fmaxf(s2.x, s2.y), fmaxf(s2.z, s2.w)),
                                 fmaxf(fmaxf(s3.x, s3.y), fmaxf(s3.z, s3.w))));
        const float rmax  = fmaxf(hmax, __shfl_xor(hmax, 16, 32));
        const float m_new = fmaxf(m_run[qh], rmax);
        const float alpha = __expf(m_run[qh] - m_new);
        m_run[qh] = m_new;

        // Rescale this half's accumulators
        #pragma unroll
        for (int g = 0; g < 8; ++g) {
          const float ag = __shfl(alpha, g + hb, 32);
          #pragma unroll
          for (int t = 0; t < 4; ++t) acc[qh * 4 + t][g] *= ag;
        }

        // exp -> bf16 A fragment (16 keys of 32, this lane's K-split)
        AFrag a;
        float psum;
        {
          const float p0 = __expf(s0.x - m_new), p1 = __expf(s0.y - m_new);
          const float p2 = __expf(s0.z - m_new), p3 = __expf(s0.w - m_new);
          const float p4 = __expf(s1.x - m_new), p5 = __expf(s1.y - m_new);
          const float p6 = __expf(s1.z - m_new), p7 = __expf(s1.w - m_new);
          const float q0_ = __expf(s2.x - m_new), q1_ = __expf(s2.y - m_new);
          const float q2_ = __expf(s2.z - m_new), q3_ = __expf(s2.w - m_new);
          const float q4_ = __expf(s3.x - m_new), q5_ = __expf(s3.y - m_new);
          const float q6_ = __expf(s3.z - m_new), q7_ = __expf(s3.w - m_new);
          psum = ((p0 + p1) + (p2 + p3)) + ((p4 + p5) + (p6 + p7))
               + ((q0_ + q1_) + (q2_ + q3_)) + ((q4_ + q5_) + (q6_ + q7_));
          a.h[ 0] = f2bf(p0);  a.h[ 1] = f2bf(p1);  a.h[ 2] = f2bf(p2);  a.h[ 3] = f2bf(p3);
          a.h[ 4] = f2bf(p4);  a.h[ 5] = f2bf(p5);  a.h[ 6] = f2bf(p6);  a.h[ 7] = f2bf(p7);
          a.h[ 8] = f2bf(q0_); a.h[ 9] = f2bf(q1_); a.h[10] = f2bf(q2_); a.h[11] = f2bf(q3_);
          a.h[12] = f2bf(q4_); a.h[13] = f2bf(q5_); a.h[14] = f2bf(q6_); a.h[15] = f2bf(q7_);
        }
        l_run[qh] = alpha * l_run[qh] + (psum + __shfl_xor(psum, 16, 32));

        // P@V: 4 WMMAs update this half's 16x64 slice
        #pragma unroll
        for (int t = 0; t < 4; ++t) {
          const int hcol = (wv << 6) + (t << 4) + nc;
          AFrag bb;
          bb.q[0] = *(const uint4*)&Vt[hcol * PV + kst];
          bb.q[1] = *(const uint4*)&Vt[hcol * PV + kst + 16];
          acc[qh * 4 + t] = __builtin_amdgcn_wmma_f32_16x16x32_bf16(
              false, a.v, false, bb.v, (short)0, acc[qh * 4 + t], false, false);
        }
      }
    }
  }

  // ---- Epilogue: normalize by l and store fp32 context ----
  {
    const int hb = (lane >> 4) << 3;
    const int n  = lane & 15;
    #pragma unroll
    for (int qh = 0; qh < 4; ++qh) {
      #pragma unroll
      for (int g = 0; g < 8; ++g) {
        const int r = (qh << 4) + g + hb;
        const float linv = 1.0f / __shfl(l_run[qh], g + hb, 32);
        #pragma unroll
        for (int t = 0; t < 4; ++t) {
          const int h = (wv << 6) + (t << 4) + n;
          Out[((size_t)b * S_ + (q0 + r)) * H_ + h] = acc[qh * 4 + t][g] * linv;
        }
      }
    }
  }
}

extern "C" void kernel_launch(void* const* d_in, const int* in_sizes, int n_in,
                              void* d_out, int out_size, void* d_ws, size_t ws_size,
                              hipStream_t stream) {
  const float* X  = (const float*)d_in[0];
  float* Out      = (float*)d_out;
  static_assert(SMEM_BYTES <= 160 * 1024, "LDS budget (2 WG per WGP)");
  (void)hipFuncSetAttribute((const void*)fattn_bf16_wmma,
                            hipFuncAttributeMaxDynamicSharedMemorySize, SMEM_BYTES);
  dim3 grid(B_ * (S_ / QT));   // 256 workgroups
  fattn_bf16_wmma<<<grid, 256, SMEM_BYTES, stream>>>(X, Out);
}
